// CellPropagation_41841571397816
// MI455X (gfx1250) — matrix operand order repaired and checked by reading, not compiled, not run on hardware
//
#include <hip/hip_runtime.h>
#include <hip/hip_bf16.h>

// ---------------------------------------------------------------------------
// MI455X (gfx1250, wave32) implementation.
// Edge MLP (204800 x [128->512->256->128]) on v_wmma_f32_16x16x32_bf16,
// M=32 rows per wave so every weight B-fragment is reused twice.
// Attention path algebraically reduced: attn = x.(Wq k) + bq.k  -> matvec.
// ---------------------------------------------------------------------------

typedef __bf16 bf16_t;
typedef bf16_t v16bf __attribute__((ext_vector_type(16)));
typedef float  v8f   __attribute__((ext_vector_type(8)));

union FragU {
    v16bf v;
    bf16_t b[16];
    uint4 q[2];
};

__device__ __forceinline__ v8f zero8() {
    v8f c = {0.f, 0.f, 0.f, 0.f, 0.f, 0.f, 0.f, 0.f};
    return c;
}

__device__ __forceinline__ v8f wmma_bf16(v16bf a, v16bf b, v8f c) {
    return __builtin_amdgcn_wmma_f32_16x16x32_bf16(
        /*neg_a=*/false, a, /*neg_b=*/false, b,
        /*c_mod=*/(short)0, c, /*reuse_a=*/false, /*reuse_b=*/false);
}

// B fragment from pre-swizzled bf16 weights: fragment f -> 512 contiguous bf16,
// lane L owns 16 contiguous elements => 2 coalesced b128 loads.
__device__ __forceinline__ v16bf load_bfrag(const bf16_t* sw, int fragIdx, int lane) {
    FragU fu;
    const uint4* p = (const uint4*)(sw + (size_t)fragIdx * 512 + lane * 16);
    fu.q[0] = p[0];
    fu.q[1] = p[1];
    return fu.v;
}

// A fragment (16x32 bf16) from row-major bf16 in LDS, row stride `stride` (elems).
__device__ __forceinline__ v16bf load_afrag_lds(const bf16_t* lds, int stride,
                                                int k0, int lane) {
    int m = lane & 15, half = lane >> 4;
    FragU fu;
    fu.q[0] = *(const uint4*)(lds + m * stride + k0 + half * 8);
    fu.q[1] = *(const uint4*)(lds + m * stride + k0 + 16 + half * 8);
    return fu.v;
}

// A fragment straight from row-major f32 in global memory, with bf16 convert.
__device__ __forceinline__ v16bf load_afrag_global(const float* base, int ld,
                                                   int k0, int lane) {
    int m = lane & 15, half = lane >> 4;
    const float* r = base + (size_t)m * ld + k0 + half * 8;
    float4 x0 = *(const float4*)(r);
    float4 x1 = *(const float4*)(r + 4);
    float4 x2 = *(const float4*)(r + 16);
    float4 x3 = *(const float4*)(r + 20);
    float t[16] = {x0.x, x0.y, x0.z, x0.w, x1.x, x1.y, x1.z, x1.w,
                   x2.x, x2.y, x2.z, x2.w, x3.x, x3.y, x3.z, x3.w};
    FragU fu;
#pragma unroll
    for (int i = 0; i < 16; ++i) fu.b[i] = (bf16_t)t[i];
    return fu.v;
}

__device__ __forceinline__ int find_seg(const int* offs, int nGraphs, int row) {
    int lo = 0, hi = nGraphs - 1;
    while (lo < hi) {
        int mid = (lo + hi + 1) >> 1;
        if (offs[mid] <= row) lo = mid; else hi = mid - 1;
    }
    return lo;
}

// ---------------------------------------------------------------------------
// Prep: convert f32 weight [K,N] (optionally transposed read) into swizzled
// bf16 B-fragment layout: dst[(kt*(N/16)+nt)*512 + lane*16 + e].
// ---------------------------------------------------------------------------
__global__ void swizzle_weight_kernel(const float* __restrict__ src,
                                      bf16_t* __restrict__ dst,
                                      int K, int N, int transposed) {
    int i = blockIdx.x * blockDim.x + threadIdx.x;
    if (i >= K * N) return;
    int k = i / N, n = i % N;
    float v = transposed ? src[(size_t)n * K + k] : src[i];
    int kt = k >> 5, kr = k & 31, hk = (kr >> 3) & 1, e = (kr & 7) + ((kr >> 4) << 3);
    int nt = n >> 4, nl = n & 15;
    int lane = hk * 16 + nl;
    dst[((size_t)(kt * (N >> 4) + nt)) * 512 + lane * 16 + e] = (bf16_t)v;
}

// Exclusive prefix sum of num_edge -> offs[0..B], single block (B <= 1024).
__global__ void scan_kernel(const int* __restrict__ num_edge, int* __restrict__ offs, int B) {
    __shared__ int s[1024];
    int tid = threadIdx.x;
    s[tid] = (tid < B) ? num_edge[tid] : 0;
    __syncthreads();
    for (int off = 1; off < 1024; off <<= 1) {
        int v = (tid >= off) ? s[tid - off] : 0;
        __syncthreads();
        s[tid] += v;
        __syncthreads();
    }
    if (tid == 0) offs[0] = 0;
    if (tid < B) offs[tid + 1] = s[tid];
}

// ---------------------------------------------------------------------------
// Edge MLP, one wave per block, M=32 rows (two 16-row sub-blocks sharing each
// weight fragment). 48KB wave-private LDS for h1/h2 bf16 (C->A transpose).
// ---------------------------------------------------------------------------
__global__ __launch_bounds__(32) void edge_mlp_kernel(
    const float* __restrict__ edge_attr, const float* __restrict__ cell,
    const bf16_t* __restrict__ W1sw, const float* __restrict__ b1,
    const bf16_t* __restrict__ W2sw, const float* __restrict__ b2,
    const bf16_t* __restrict__ W3sw, const float* __restrict__ b3,
    const int* __restrict__ offs, int nGraphs,
    float* __restrict__ out_edge, int totalE) {
    __shared__ bf16_t h1[32 * 512];  // 32KB
    __shared__ bf16_t h2[32 * 256];  // 16KB

    int lane = threadIdx.x & 31;
    int rowBase = blockIdx.x * 32;
    if (rowBase >= totalE) return;
    int n = lane & 15, half = lane >> 4;

    // A1: full K=128 of both 16-row sub-blocks, in registers (8 fragments).
    v16bf a1[2][4];
#pragma unroll
    for (int s = 0; s < 2; ++s)
#pragma unroll
        for (int kt = 0; kt < 4; ++kt)
            a1[s][kt] = load_afrag_global(edge_attr + (size_t)(rowBase + 16 * s) * 128,
                                          128, kt * 32, lane);

    // ---- layer 1: 32x128 @ 128x512 ----
    for (int nt = 0; nt < 32; ++nt) {
        v8f c0 = zero8(), c1 = zero8();
#pragma unroll
        for (int kt = 0; kt < 4; ++kt) {
            v16bf bf = load_bfrag(W1sw, kt * 32 + nt, lane);
            c0 = wmma_bf16(a1[0][kt], bf, c0);
            c1 = wmma_bf16(a1[1][kt], bf, c1);
        }
        float bias = b1[nt * 16 + n];
#pragma unroll
        for (int r = 0; r < 8; ++r) {
            int rr = r + half * 8;
            h1[rr * 512 + nt * 16 + n]        = (bf16_t)fmaxf(c0[r] + bias, 0.f);
            h1[(16 + rr) * 512 + nt * 16 + n] = (bf16_t)fmaxf(c1[r] + bias, 0.f);
        }
    }

    // ---- layer 2: 32x512 @ 512x256 ----
    for (int nt = 0; nt < 16; ++nt) {
        if (nt + 1 < 16)  // prefetch next weight slab (gfx1250 global_prefetch)
            __builtin_prefetch(W2sw + (size_t)(nt + 1) * 512, 0, 0);
        v8f c0 = zero8(), c1 = zero8();
#pragma unroll
        for (int kt = 0; kt < 16; ++kt) {
            v16bf bf = load_bfrag(W2sw, kt * 16 + nt, lane);
            c0 = wmma_bf16(load_afrag_lds(h1, 512, kt * 32, lane), bf, c0);
            c1 = wmma_bf16(load_afrag_lds(h1 + 16 * 512, 512, kt * 32, lane), bf, c1);
        }
        float bias = b2[nt * 16 + n];
#pragma unroll
        for (int r = 0; r < 8; ++r) {
            int rr = r + half * 8;
            h2[rr * 256 + nt * 16 + n]        = (bf16_t)fmaxf(c0[r] + bias, 0.f);
            h2[(16 + rr) * 256 + nt * 16 + n] = (bf16_t)fmaxf(c1[r] + bias, 0.f);
        }
    }

    // graph segment of each row this lane stores (16 rows: rr and rr+16)
    int segs[16];
#pragma unroll
    for (int r = 0; r < 8; ++r) {
        segs[r]     = find_seg(offs, nGraphs, rowBase + r + half * 8);
        segs[r + 8] = find_seg(offs, nGraphs, rowBase + 16 + r + half * 8);
    }

    // ---- layer 3: 32x256 @ 256x128 (+ b3 + cell[seg]) ----
    for (int nt = 0; nt < 8; ++nt) {
        v8f c0 = zero8(), c1 = zero8();
#pragma unroll
        for (int kt = 0; kt < 8; ++kt) {
            v16bf bf = load_bfrag(W3sw, kt * 8 + nt, lane);
            c0 = wmma_bf16(load_afrag_lds(h2, 256, kt * 32, lane), bf, c0);
            c1 = wmma_bf16(load_afrag_lds(h2 + 16 * 256, 256, kt * 32, lane), bf, c1);
        }
        int col = nt * 16 + n;
        float bias = b3[col];
#pragma unroll
        for (int r = 0; r < 8; ++r) {
            int row0 = rowBase + r + half * 8;
            int row1 = row0 + 16;
            out_edge[((size_t)row0 << 7) + col] =
                c0[r] + bias + cell[((size_t)segs[r] << 7) + col];
            out_edge[((size_t)row1 << 7) + col] =
                c1[r] + bias + cell[((size_t)segs[r + 8] << 7) + col];
        }
    }
}

// ---------------------------------------------------------------------------
// K = cell@Wk+bk, V = cell@Wv+bv, T = K@Wq^T  (per 16-row strip, WMMA).
// ---------------------------------------------------------------------------
__global__ __launch_bounds__(64) void kv_kernel(
    const float* __restrict__ cell,
    const bf16_t* __restrict__ Wksw, const float* __restrict__ bk,
    const bf16_t* __restrict__ Wvsw, const float* __restrict__ bv,
    const bf16_t* __restrict__ WqTsw,
    float* __restrict__ Kout, float* __restrict__ Vout, float* __restrict__ Tout, int B) {
    __shared__ bf16_t kbuf[2][16 * 128];
    int wave = threadIdx.x >> 5, lane = threadIdx.x & 31;
    int rowBase = (blockIdx.x * 2 + wave) * 16;
    if (rowBase >= B) return;
    int n = lane & 15, half = lane >> 4;

    v16bf a[4];
#pragma unroll
    for (int kt = 0; kt < 4; ++kt)
        a[kt] = load_afrag_global(cell + (size_t)rowBase * 128, 128, kt * 32, lane);

    // K (also staged to LDS bf16 as A-matrix input of T)
    for (int nt = 0; nt < 8; ++nt) {
        v8f c = zero8();
#pragma unroll
        for (int kt = 0; kt < 4; ++kt)
            c = wmma_bf16(a[kt], load_bfrag(Wksw, kt * 8 + nt, lane), c);
        float bias = bk[nt * 16 + n];
#pragma unroll
        for (int r = 0; r < 8; ++r) {
            float f = c[r] + bias;
            int row = rowBase + r + half * 8;
            Kout[((size_t)row << 7) + nt * 16 + n] = f;
            kbuf[wave][(r + half * 8) * 128 + nt * 16 + n] = (bf16_t)f;
        }
    }
    // V
    for (int nt = 0; nt < 8; ++nt) {
        v8f c = zero8();
#pragma unroll
        for (int kt = 0; kt < 4; ++kt)
            c = wmma_bf16(a[kt], load_bfrag(Wvsw, kt * 8 + nt, lane), c);
        float bias = bv[nt * 16 + n];
#pragma unroll
        for (int r = 0; r < 8; ++r) {
            int row = rowBase + r + half * 8;
            Vout[((size_t)row << 7) + nt * 16 + n] = c[r] + bias;
        }
    }
    // T = K @ Wq^T
    v16bf a2[4];
#pragma unroll
    for (int kt = 0; kt < 4; ++kt)
        a2[kt] = load_afrag_lds(kbuf[wave], 128, kt * 32, lane);
    for (int nt = 0; nt < 8; ++nt) {
        v8f c = zero8();
#pragma unroll
        for (int kt = 0; kt < 4; ++kt)
            c = wmma_bf16(a2[kt], load_bfrag(WqTsw, kt * 8 + nt, lane), c);
#pragma unroll
        for (int r = 0; r < 8; ++r) {
            int row = rowBase + r + half * 8;
            Tout[((size_t)row << 7) + nt * 16 + n] = c[r];
        }
    }
}

// ---------------------------------------------------------------------------
// Attention + residual: bandwidth-bound node path, one block per graph.
// ---------------------------------------------------------------------------
__device__ __forceinline__ float wred_sum(float v) {
#pragma unroll
    for (int o = 16; o > 0; o >>= 1) v += __shfl_xor(v, o, 32);
    return v;
}
__device__ __forceinline__ float wred_max(float v) {
#pragma unroll
    for (int o = 16; o > 0; o >>= 1) v = fmaxf(v, __shfl_xor(v, o, 32));
    return v;
}

__global__ __launch_bounds__(128) void attn_kernel(
    const float* __restrict__ x, const float* __restrict__ cell,
    const unsigned char* __restrict__ mask,
    const float* __restrict__ bq, const float* __restrict__ param,
    const float* __restrict__ Kws, const float* __restrict__ Vws,
    const float* __restrict__ Tws, float* __restrict__ newx, int N) {
    int b = blockIdx.x;
    int tid = threadIdx.x, lane = tid & 31, wave = tid >> 5;
    __shared__ float tS[128], vS[128], cS[128];
    __shared__ float attnS[128], gateS[128];
    __shared__ float red[8];
    __shared__ float sB;

    tS[tid] = Tws[((size_t)b << 7) + tid];
    vS[tid] = Vws[((size_t)b << 7) + tid];
    cS[tid] = cell[((size_t)b << 7) + tid];
    if (wave == 0) {  // sB = bq . K_b
        float p = 0.f;
        for (int c = lane; c < 128; c += 32) p += bq[c] * Kws[((size_t)b << 7) + c];
        p = wred_sum(p);
        if (lane == 0) sB = p;
    }
    __syncthreads();

    const float* xg = x + (size_t)b * N * 128;
    for (int nr = wave; nr < N; nr += 4) {  // logits: x_n . t + sB
        float4 xv = *(const float4*)(xg + (size_t)nr * 128 + lane * 4);
        float p = xv.x * tS[lane * 4] + xv.y * tS[lane * 4 + 1] +
                  xv.z * tS[lane * 4 + 2] + xv.w * tS[lane * 4 + 3];
        p = wred_sum(p);
        if (lane == 0) attnS[nr] = p + sB;
    }
    __syncthreads();

    // masked softmax over N, scaled by 1/sqrt(128)
    int msk = 0;
    float val = -3.4e38f;
    if (tid < N) {
        msk = mask[(size_t)b * N + tid];
        val = msk ? -1e9f : attnS[tid];
    }
    float wm = wred_max(val);
    if (lane == 0) red[wave] = wm;
    __syncthreads();
    float bmax = fmaxf(fmaxf(red[0], red[1]), fmaxf(red[2], red[3]));
    float e = (tid < N) ? __expf(val - bmax) : 0.f;
    float ws = wred_sum(e);
    if (lane == 0) red[4 + wave] = ws;
    __syncthreads();
    float bsum = red[4] + red[5] + red[6] + red[7];
    if (tid < N) {
        attnS[tid] = e / bsum * 0.08838834764831845f;  // /sqrt(128)
        gateS[tid] = msk ? 0.f : param[tid];
    }
    __syncthreads();

    float* og = newx + (size_t)b * N * 128;
    for (int nr = wave; nr < N; nr += 4) {
        float a = attnS[nr], g = gateS[nr];
        int c = lane * 4;
        float4 xv = *(const float4*)(xg + (size_t)nr * 128 + c);
        float4 o;
        o.x = a * vS[c + 0] + g * cS[c + 0] + xv.x;
        o.y = a * vS[c + 1] + g * cS[c + 1] + xv.y;
        o.z = a * vS[c + 2] + g * cS[c + 2] + xv.z;
        o.w = a * vS[c + 3] + g * cS[c + 3] + xv.w;
        *(float4*)(og + (size_t)nr * 128 + c) = o;
    }
}

// ---------------------------------------------------------------------------
extern "C" void kernel_launch(void* const* d_in, const int* in_sizes, int n_in,
                              void* d_out, int out_size, void* d_ws, size_t ws_size,
                              hipStream_t stream) {
    const int D = 128, H1 = 512, H2 = 256;
    const float* x         = (const float*)d_in[0];
    const float* cell      = (const float*)d_in[1];
    const unsigned char* mask = (const unsigned char*)d_in[2];
    const float* edge_attr = (const float*)d_in[3];
    const int*   num_edge  = (const int*)d_in[4];
    const float* Wq = (const float*)d_in[5];
    const float* bq = (const float*)d_in[6];
    const float* Wk = (const float*)d_in[7];
    const float* bk = (const float*)d_in[8];
    const float* Wv = (const float*)d_in[9];
    const float* bv = (const float*)d_in[10];
    const float* W1 = (const float*)d_in[11];
    const float* b1 = (const float*)d_in[12];
    const float* W2 = (const float*)d_in[13];
    const float* b2 = (const float*)d_in[14];
    const float* W3 = (const float*)d_in[15];
    const float* b3 = (const float*)d_in[16];
    const float* param = (const float*)d_in[17];

    const int B = in_sizes[1] / D;
    const int BN = in_sizes[0] / D;
    const int N = BN / B;
    const int TOTAL_E = in_sizes[3] / D;

    // workspace layout
    char* wsb = (char*)d_ws;
    size_t off = 0;
    auto alloc = [&](size_t bytes) { size_t r = off; off = (off + bytes + 255) & ~(size_t)255; return r; };
    bf16_t* W1sw  = (bf16_t*)(wsb + alloc((size_t)D * H1 * 2));
    bf16_t* W2sw  = (bf16_t*)(wsb + alloc((size_t)H1 * H2 * 2));
    bf16_t* W3sw  = (bf16_t*)(wsb + alloc((size_t)H2 * D * 2));
    bf16_t* Wksw  = (bf16_t*)(wsb + alloc((size_t)D * D * 2));
    bf16_t* Wvsw  = (bf16_t*)(wsb + alloc((size_t)D * D * 2));
    bf16_t* WqTsw = (bf16_t*)(wsb + alloc((size_t)D * D * 2));
    int*   offs = (int*)(wsb + alloc((size_t)(B + 1) * 4));
    float* Kws  = (float*)(wsb + alloc((size_t)B * D * 4));
    float* Vws  = (float*)(wsb + alloc((size_t)B * D * 4));
    float* Tws  = (float*)(wsb + alloc((size_t)B * D * 4));
    (void)ws_size; (void)n_in; (void)out_size;

    float* newx     = (float*)d_out;
    float* out_edge = newx + (size_t)BN * D;

    // 1) weight prep (bf16 + WMMA-B swizzle)
    swizzle_weight_kernel<<<(D * H1 + 255) / 256, 256, 0, stream>>>(W1, W1sw, D, H1, 0);
    swizzle_weight_kernel<<<(H1 * H2 + 255) / 256, 256, 0, stream>>>(W2, W2sw, H1, H2, 0);
    swizzle_weight_kernel<<<(H2 * D + 255) / 256, 256, 0, stream>>>(W3, W3sw, H2, D, 0);
    swizzle_weight_kernel<<<(D * D + 255) / 256, 256, 0, stream>>>(Wk, Wksw, D, D, 0);
    swizzle_weight_kernel<<<(D * D + 255) / 256, 256, 0, stream>>>(Wv, Wvsw, D, D, 0);
    swizzle_weight_kernel<<<(D * D + 255) / 256, 256, 0, stream>>>(Wq, WqTsw, D, D, 1);

    // 2) edge segment offsets
    scan_kernel<<<1, 1024, 0, stream>>>(num_edge, offs, B);

    // 3) K, V, T (WMMA)
    int kvStrips = (B + 15) / 16;
    kv_kernel<<<(kvStrips + 1) / 2, 64, 0, stream>>>(cell, Wksw, bk, Wvsw, bv, WqTsw,
                                                     Kws, Vws, Tws, B);

    // 4) edge MLP (dominant WMMA work), 32 rows per wave
    int eBlocks = (TOTAL_E + 31) / 32;
    edge_mlp_kernel<<<eBlocks, 32, 0, stream>>>(
        edge_attr, cell, W1sw, b1, W2sw, b2, W3sw, b3, offs, B, out_edge, TOTAL_E);

    // 5) attention + residual (bandwidth-bound)
    attn_kernel<<<B, 128, 0, stream>>>(x, cell, mask, bq, param, Kws, Vws, Tws, newx, N);
}
